// GRAPH_AUTOENCODER_20255065768572
// MI455X (gfx1250) — compile-verified, matrix operand order checked by reading, not run on hardware
//
#include <hip/hip_runtime.h>
#include <hip/hip_bf16.h>

// ---------------------------------------------------------------------------
// CDNA5 (gfx1250) wave32 WMMA implementation of the graph autoencoder.
// All matmuls -> v_wmma_f32_16x16x32_bf16 (f32 accumulate).
//  * all fragment loads K-contiguous per lane -> guaranteed b128 loads
//  * 2x2 register blocking: each wave computes a 32x32 C block
//  * row-major outputs computed as C^T tiles (operand swap) so each lane
//    owns 8 contiguous output elements -> vector stores, vector epilogue
// ---------------------------------------------------------------------------

typedef __bf16 bf16_t;
typedef __attribute__((ext_vector_type(16))) __bf16 v16bf;
typedef __attribute__((ext_vector_type(8)))  __bf16 v8bf;
typedef __attribute__((ext_vector_type(8)))  float  v8f;
typedef __attribute__((ext_vector_type(4)))  float  v4f;

static constexpr int GG  = 256;   // graphs
static constexpr int NN  = 128;   // nodes per graph
static constexpr int FF  = 64;    // input features
static constexpr int HH1 = 256;
static constexpr int HH2 = 128;

#define BN_SCALE 0.99999500003749975f   /* 1/sqrt(1+1e-5) */

enum {
  F_BIAS        = 1,
  F_BN          = 2,
  F_RELU        = 4,
  F_ADDR        = 8,     // += residual (bf16, per-graph) before final relu
  F_RELU_FIRST  = 16,    // relu BEFORE bn (feature decoder path)
  F_SIGMOID     = 32,
  F_THRESH      = 64,    // also write (val>0.5) to C2 (bf16, ld=128)
  F_OUTF        = 128,   // write f32 output Cf
  F_OUTB        = 256,   // write bf16 output Cb
  F_STORE_TRANS = 512    // Cb stored transposed: Cb[col*ldcb + row]
};

// --- small vector helpers ---------------------------------------------------
__device__ __forceinline__ v8f load8f(const float* p) {
  v4f a = *(const v4f*)p, b = *(const v4f*)(p + 4);
  v8f r;
#pragma unroll
  for (int j = 0; j < 4; ++j) { r[j] = a[j]; r[j + 4] = b[j]; }
  return r;
}
__device__ __forceinline__ void store8f(float* p, v8f v) {
  v4f a, b;
#pragma unroll
  for (int j = 0; j < 4; ++j) { a[j] = v[j]; b[j] = v[j + 4]; }
  *(v4f*)p = a; *(v4f*)(p + 4) = b;
}
__device__ __forceinline__ void store8bf(bf16_t* p, v8f v) {
  v8bf o;
#pragma unroll
  for (int j = 0; j < 8; ++j) o[j] = (bf16_t)v[j];
  *(v8bf*)p = o;
}
__device__ __forceinline__ v8f load8bf_f(const bf16_t* p) {
  v8bf x = *(const v8bf*)p;
  v8f r;
#pragma unroll
  for (int j = 0; j < 8; ++j) r[j] = (float)x[j];
  return r;
}

// --- unified row-major fragment loader (ISA 7.12.2 K-grouping) --------------
// Operand stored [R x K] row-major; lane L supplies row (L&15); lane-half
// selects K+8.  Two contiguous 8-element groups -> explicit vector loads.
template <typename T>
__device__ __forceinline__ v16bf load_frag_rm(const T* __restrict__ X, int ld,
                                              int r0, int k0, int lane) {
  const T* p = X + (size_t)(r0 + (lane & 15)) * ld + k0 + ((lane >> 4) << 3);
  v16bf v;
  if constexpr (sizeof(T) == 2) {
    v8bf lo = *(const v8bf*)p;
    v8bf hi = *(const v8bf*)(p + 16);
#pragma unroll
    for (int j = 0; j < 8; ++j) { v[j] = lo[j]; v[j + 8] = hi[j]; }
  } else {
    v4f q0 = *(const v4f*)p,        q1 = *(const v4f*)(p + 4);
    v4f q2 = *(const v4f*)(p + 16), q3 = *(const v4f*)(p + 20);
#pragma unroll
    for (int j = 0; j < 4; ++j) {
      v[j]      = (bf16_t)q0[j];  v[4 + j]  = (bf16_t)q1[j];
      v[8 + j]  = (bf16_t)q2[j];  v[12 + j] = (bf16_t)q3[j];
    }
  }
  return v;
}

__device__ __forceinline__ v8f wmma_bf16(v16bf a, v16bf b, v8f c) {
  return __builtin_amdgcn_wmma_f32_16x16x32_bf16(false, a, false, b, (short)0,
                                                 c, false, false);
}

// --- Generic per-graph GEMM: C[g] = epilogue(A[g] (128xK) * B[g]^T) ---------
// A: [128 x K] row-major (TA).  B: [N x K] row-major (TB) == B^T of math B.
template <typename TA, typename TB, int N, int K, int FLAGS>
__global__ void __launch_bounds__(256)
gemm128(const TA* __restrict__ Ag, long long sAg, int lda,
        const TB* __restrict__ Bg, long long sBg, int ldb,
        const float* __restrict__ bias, const float* __restrict__ gamma,
        const float* __restrict__ beta,
        const bf16_t* __restrict__ Rg, long long sRg, int ldr,
        float*  __restrict__ Cf, long long sCfg, int ldcf,
        bf16_t* __restrict__ Cb, long long sCbg, int ldcb,
        bf16_t* __restrict__ C2, long long sC2g) {
  constexpr bool TRS = (FLAGS & F_STORE_TRANS) != 0;
  const int g    = blockIdx.x;
  const int lane = threadIdx.x & 31;
  const int wave = threadIdx.x >> 5;
  const TA* A = Ag + (size_t)g * sAg;
  const TB* B = Bg + (size_t)g * sBg;

  // vectorized row-major epilogue: lane owns 8 contiguous columns of one row
  auto epi = [&](v8f v, int row, int cb) {
    if (FLAGS & F_BIAS) v += load8f(bias + cb);
    if (FLAGS & F_RELU_FIRST)
#pragma unroll
      for (int j = 0; j < 8; ++j) v[j] = fmaxf(v[j], 0.0f);
    if (FLAGS & F_BN)
      v = load8f(gamma + cb) * (v * BN_SCALE) + load8f(beta + cb);
    if (FLAGS & F_ADDR)
      v += load8bf_f(Rg + (size_t)g * sRg + (size_t)row * ldr + cb);
    if (FLAGS & F_RELU)
#pragma unroll
      for (int j = 0; j < 8; ++j) v[j] = fmaxf(v[j], 0.0f);
    if (FLAGS & F_SIGMOID)
#pragma unroll
      for (int j = 0; j < 8; ++j) v[j] = 1.0f / (1.0f + __expf(-v[j]));
    if (FLAGS & F_OUTF)
      store8f(Cf + (size_t)g * sCfg + (size_t)row * ldcf + cb, v);
    if (FLAGS & F_OUTB)
      store8bf(Cb + (size_t)g * sCbg + (size_t)row * ldcb + cb, v);
    if (FLAGS & F_THRESH) {
      v8f t;
#pragma unroll
      for (int j = 0; j < 8; ++j) t[j] = v[j] > 0.5f ? 1.0f : 0.0f;
      store8bf(C2 + (size_t)g * sC2g + (size_t)row * 128 + cb, t);
    }
  };

  constexpr int NB = 4 * (N >> 5);          // 32x32 blocks (M=128 -> 4)
  for (int t = wave; t < NB; t += 8) {      // wave-uniform: EXEC all-1s
    const int tm = (t & 3) << 5;
    const int tn = (t >> 2) << 5;
    v8f c00 = {}, c01 = {}, c10 = {}, c11 = {};
#pragma unroll
    for (int k0 = 0; k0 < K; k0 += 32) {
      if (k0 + 32 < K) {  // compile-time (K is template)
        __builtin_prefetch(A + (size_t)(tm + (lane & 15)) * lda + k0 + 32, 0, 3);
        __builtin_prefetch(B + (size_t)(tn + (lane & 15)) * ldb + k0 + 32, 0, 3);
      }
      v16bf a0 = load_frag_rm<TA>(A, lda, tm,      k0, lane);
      v16bf a1 = load_frag_rm<TA>(A, lda, tm + 16, k0, lane);
      v16bf b0 = load_frag_rm<TB>(B, ldb, tn,      k0, lane);
      v16bf b1 = load_frag_rm<TB>(B, ldb, tn + 16, k0, lane);
      if (TRS) {            // natural orientation: lane = C-col
        c00 = wmma_bf16(a0, b0, c00);
        c01 = wmma_bf16(a0, b1, c01);
        c10 = wmma_bf16(a1, b0, c10);
        c11 = wmma_bf16(a1, b1, c11);
      } else {              // swapped: D' = C^T tile, lane = C-row
        c00 = wmma_bf16(b0, a0, c00);
        c01 = wmma_bf16(b1, a0, c01);
        c10 = wmma_bf16(b0, a1, c10);
        c11 = wmma_bf16(b1, a1, c11);
      }
    }
    if (TRS) {
      // lane = C-col, vgprs = 8 consecutive C-rows -> contiguous in C^T buf
      const int rb = (lane >> 4) << 3;
      auto st = [&](v8f v, int tmi, int tni) {
        store8bf(Cb + (size_t)g * sCbg +
                     (size_t)(tni + (lane & 15)) * ldcb + tmi + rb, v);
      };
      st(c00, tm, tn);      st(c01, tm, tn + 16);
      st(c10, tm + 16, tn); st(c11, tm + 16, tn + 16);
    } else {
      const int row = lane & 15;
      const int cb  = (lane >> 4) << 3;
      epi(c00, tm + row,      tn + cb);
      epi(c01, tm + row,      tn + 16 + cb);
      epi(c10, tm + 16 + row, tn + cb);
      epi(c11, tm + 16 + row, tn + 16 + cb);
    }
  }
}

// --- weight transpose + f32->bf16: in [K x N] -> out [N x K] ----------------
__global__ void __launch_bounds__(256)
transpose_bf16_k(const float* __restrict__ in, bf16_t* __restrict__ out,
                 int K, int N) {
  int idx = blockIdx.x * 256 + threadIdx.x;
  if (idx < K * N) {
    int k = idx / N, n = idx - k * N;
    out[(size_t)n * K + k] = (bf16_t)in[idx];
  }
}

// --- D^-1/2 (A + I) D^-1/2 -> bf16 ------------------------------------------
template <typename T>
__global__ void __launch_bounds__(128)
gcn_norm_k(const T* __restrict__ adjg, long long sg, int lda,
           bf16_t* __restrict__ Ab, long long sAb) {
  __shared__ float dinv[NN];
  const int g = blockIdx.x, i = threadIdx.x;
  const T* adj = adjg + (size_t)g * sg;
  float s = 1.0f;  // self loop
  for (int j = 0; j < NN; ++j) s += (float)adj[(size_t)i * lda + j];
  dinv[i] = rsqrtf(fmaxf(s, 1e-12f));
  __syncthreads();
  const float di = dinv[i];
  bf16_t* out = Ab + (size_t)g * sAb + (size_t)i * NN;
  for (int j = 0; j < NN; ++j) {
    float a = (float)adj[(size_t)i * lda + j] + (i == j ? 1.0f : 0.0f);
    out[j] = (bf16_t)(a * di * dinv[j]);
  }
}

// --- per-row L2 normalize (z) -----------------------------------------------
__global__ void __launch_bounds__(128)
rownorm_k(const bf16_t* __restrict__ Zb, long long sZb, int ldz,
          float* __restrict__ O1, long long sO1,
          float* __restrict__ O2, long long sO2) {
  const int g = blockIdx.x, i = threadIdx.x;
  const bf16_t* zr = Zb + (size_t)g * sZb + (size_t)i * ldz;
  float ss = 0.0f;
  for (int j = 0; j < HH2; ++j) { float v = (float)zr[j]; ss += v * v; }
  const float inv = 1.0f / fmaxf(sqrtf(ss), 1e-12f);
  for (int j = 0; j < HH2; ++j) {
    float v = (float)zr[j] * inv;
    O1[(size_t)g * sO1 + (size_t)i * HH2 + j] = v;
    if (O2) O2[(size_t)g * sO2 + (size_t)i * HH2 + j] = v;
  }
}

// --- global max pool over nodes (optionally z + scale*noise) ----------------
__global__ void __launch_bounds__(128)
maxpool_k(const float* __restrict__ Zg, long long sZ, int ldz,
          const float* __restrict__ noise, long long sN, float nscale,
          float* __restrict__ O1, float* __restrict__ O2) {
  const int g = blockIdx.x, h = threadIdx.x;
  const float* z  = Zg + (size_t)g * sZ;
  const float* nz = noise ? noise + (size_t)g * sN : nullptr;
  float m = -3.402823466e38f;
  for (int r = 0; r < NN; ++r) {
    float v = z[(size_t)r * ldz + h];
    if (nz) v += nscale * nz[(size_t)r * HH2 + h];
    m = fmaxf(m, v);
  }
  O1[(size_t)g * HH2 + h] = m;
  if (O2) O2[(size_t)g * HH2 + h] = m;
}

// ---------------------------------------------------------------------------
// host side
// ---------------------------------------------------------------------------
template <typename TA, typename TB, int N, int K, int FLAGS>
static void launch_gemm(int grid, hipStream_t st,
                        const TA* A, long long sA, int lda,
                        const TB* B, long long sB, int ldb,
                        const float* bias = nullptr, const float* gamma = nullptr,
                        const float* beta = nullptr,
                        const bf16_t* R = nullptr, long long sR = 0, int ldr = 0,
                        float* Cf = nullptr, long long sCf = 0, int ldcf = 0,
                        bf16_t* Cb = nullptr, long long sCb = 0, int ldcb = 0,
                        bf16_t* C2 = nullptr, long long sC2 = 0) {
  gemm128<TA, TB, N, K, FLAGS><<<grid, 256, 0, st>>>(
      A, sA, lda, B, sB, ldb, bias, gamma, beta, R, sR, ldr,
      Cf, sCf, ldcf, Cb, sCb, ldcb, C2, sC2);
}

struct BlockW {  // transposed bf16 weights + f32 vectors for one res-block
  const bf16_t *W1t, *W2t, *Wst;
  const float *b1, *b2, *g1, *be1, *g2, *be2, *bs;
};

// Two-block GCN residual encoder + row-normalize.  Input x: [G,128,64] f32.
static void run_encode(const float* xin, long long sx,
                       const bf16_t* Abf, bf16_t* T0, bf16_t* T1, bf16_t* T2,
                       const BlockW& B0, const BlockW& B1,
                       float* O1, float* O2, hipStream_t st) {
  const long long sT = (long long)NN * HH1;  // 32768 elems / graph slot
  const long long sA = (long long)NN * NN;
  // ---- block 0: 64 -> 256
  launch_gemm<float, bf16_t, HH1, FF, F_BIAS | F_OUTB>(
      GG, st, xin, sx, FF, B0.Wst, 0, FF, B0.bs, nullptr, nullptr,
      nullptr, 0, 0, nullptr, 0, 0, T0, sT, HH1);                          // res
  launch_gemm<float, bf16_t, HH1, FF, F_OUTB | F_STORE_TRANS>(
      GG, st, xin, sx, FF, B0.W1t, 0, FF, nullptr, nullptr, nullptr,
      nullptr, 0, 0, nullptr, 0, 0, T1, sT, NN);                           // t^T
  launch_gemm<bf16_t, bf16_t, HH1, NN, F_BIAS | F_BN | F_RELU | F_OUTB>(
      GG, st, Abf, sA, NN, T1, sT, NN, B0.b1, B0.g1, B0.be1,
      nullptr, 0, 0, nullptr, 0, 0, T2, sT, HH1);                          // h
  launch_gemm<bf16_t, bf16_t, HH1, HH1, F_OUTB | F_STORE_TRANS>(
      GG, st, T2, sT, HH1, B0.W2t, 0, HH1, nullptr, nullptr, nullptr,
      nullptr, 0, 0, nullptr, 0, 0, T1, sT, NN);                           // t2^T
  launch_gemm<bf16_t, bf16_t, HH1, NN,
              F_BIAS | F_BN | F_ADDR | F_RELU | F_OUTB>(
      GG, st, Abf, sA, NN, T1, sT, NN, B0.b2, B0.g2, B0.be2,
      T0, sT, HH1, nullptr, 0, 0, T2, sT, HH1);                            // x1
  // ---- block 1: 256 -> 128
  launch_gemm<bf16_t, bf16_t, HH2, HH1, F_BIAS | F_OUTB>(
      GG, st, T2, sT, HH1, B1.Wst, 0, HH1, B1.bs, nullptr, nullptr,
      nullptr, 0, 0, nullptr, 0, 0, T0, sT, HH2);                          // res2
  launch_gemm<bf16_t, bf16_t, HH2, HH1, F_OUTB | F_STORE_TRANS>(
      GG, st, T2, sT, HH1, B1.W1t, 0, HH1, nullptr, nullptr, nullptr,
      nullptr, 0, 0, nullptr, 0, 0, T1, sT, NN);                           // t^T
  launch_gemm<bf16_t, bf16_t, HH2, NN, F_BIAS | F_BN | F_RELU | F_OUTB>(
      GG, st, Abf, sA, NN, T1, sT, NN, B1.b1, B1.g1, B1.be1,
      nullptr, 0, 0, nullptr, 0, 0, T2, sT, HH2);                          // h
  launch_gemm<bf16_t, bf16_t, HH2, HH2, F_OUTB | F_STORE_TRANS>(
      GG, st, T2, sT, HH2, B1.W2t, 0, HH2, nullptr, nullptr, nullptr,
      nullptr, 0, 0, nullptr, 0, 0, T1, sT, NN);                           // t2^T
  launch_gemm<bf16_t, bf16_t, HH2, NN,
              F_BIAS | F_BN | F_ADDR | F_RELU | F_OUTB>(
      GG, st, Abf, sA, NN, T1, sT, NN, B1.b2, B1.g2, B1.be2,
      T0, sT, HH2, nullptr, 0, 0, T2, sT, HH2);                            // z0
  rownorm_k<<<GG, 128, 0, st>>>(T2, sT, HH2, O1, (long long)NN * HH2, O2,
                                (long long)NN * HH2);
}

extern "C" void kernel_launch(void* const* d_in, const int* in_sizes, int n_in,
                              void* d_out, int out_size, void* d_ws,
                              size_t ws_size, hipStream_t stream) {
  (void)in_sizes; (void)n_in; (void)out_size; (void)ws_size;
  const float* x     = (const float*)d_in[0];
  const float* adj   = (const float*)d_in[1];
  const float* noise = (const float*)d_in[2];
  const float* p[34];
  for (int i = 0; i < 34; ++i) p[i] = (const float*)d_in[i];
  const float *bf1 = p[25], *gf = p[26], *bef = p[27], *bf2 = p[29],
              *bp1 = p[31], *bp2 = p[33];

  // ---- output slot offsets (floats), tuple order --------------------------
  float* out = (float*)d_out;
  const long long o_z       = 4194304;
  const long long o_zg      = 8388608;
  const long long o_xrec    = 8421376;
  const long long o_adjrec  = 10518528;
  const long long o_ztilde  = 14712832;
  const long long o_ztildeg = 18907136;
  const long long o_zgmlp   = 18939904;  // z_prime_g_mlp follows contiguously

  // ---- workspace carve-up --------------------------------------------------
  bf16_t* Abf = (bf16_t*)d_ws;                        // G*128*128 bf16
  bf16_t* T0  = Abf + (size_t)GG * NN * NN;           // G*128*256 bf16 each
  bf16_t* T1  = T0 + (size_t)GG * NN * HH1;
  bf16_t* T2  = T1 + (size_t)GG * NN * HH1;
  float*  Zf  = (float*)(T2 + (size_t)GG * NN * HH1); // G*128*128 f32
  float*  pooled = Zf + (size_t)GG * NN * HH2;        // 512*128 f32
  bf16_t* wt = (bf16_t*)(pooled + 512 * HH2);         // transposed weights

  struct { int idx, K, N; } wl[11] = {
      {3, FF, HH1},   {5, HH1, HH1},  {11, FF, HH1},   // W1_0, W2_0, Ws_0
      {13, HH1, HH2}, {15, HH2, HH2}, {21, HH1, HH2},  // W1_1, W2_1, Ws_1
      {23, HH2, HH2},                                  // Wed
      {24, HH2, HH1}, {28, HH1, FF},                   // Wf1, Wf2
      {30, HH2, HH2}, {32, HH2, HH2}};                 // Wp1, Wp2
  bf16_t* wtp[11];
  {
    bf16_t* cur = wt;
    for (int i = 0; i < 11; ++i) {
      wtp[i] = cur;
      int sz = wl[i].K * wl[i].N;
      transpose_bf16_k<<<(sz + 255) / 256, 256, 0, stream>>>(p[wl[i].idx], cur,
                                                             wl[i].K, wl[i].N);
      cur += sz;
    }
  }
  BlockW B0{wtp[0], wtp[1], wtp[2], p[4], p[6], p[7], p[8], p[9], p[10], p[12]};
  BlockW B1{wtp[3], wtp[4], wtp[5], p[14], p[16], p[17], p[18], p[19], p[20], p[22]};
  const bf16_t *Wedt = wtp[6], *Wf1t = wtp[7], *Wf2t = wtp[8], *Wp1t = wtp[9],
               *Wp2t = wtp[10];

  const long long sA = (long long)NN * NN;    // 16384
  const long long sT = (long long)NN * HH1;   // 32768
  const long long sZ = (long long)NN * HH2;   // 16384
  const long long sX = (long long)NN * FF;    // 8192

  // (1) adj passthrough
  hipMemcpyAsync(d_out, adj, (size_t)GG * NN * NN * sizeof(float),
                 hipMemcpyDeviceToDevice, stream);

  // (2) A_hat = D^-1/2 (adj + I) D^-1/2
  gcn_norm_k<float><<<GG, 128, 0, stream>>>(adj, sA, NN, Abf, sA);

  // (3) z = encode(x, A_hat)  -> d_out z slot + Zf
  run_encode(x, sX, Abf, T0, T1, T2, B0, B1, out + o_z, Zf, stream);

  // (4) ZW = z @ Wed  (bf16 in T0)
  launch_gemm<float, bf16_t, HH2, HH2, F_OUTB>(
      GG, stream, Zf, sZ, HH2, Wedt, 0, HH2, nullptr, nullptr, nullptr,
      nullptr, 0, 0, nullptr, 0, 0, T0, sT, HH2);
  // (5) adj_recon = sigmoid(ZW @ z^T); adj_new = (adj_recon > 0.5) -> T1
  launch_gemm<bf16_t, float, NN, HH2, F_SIGMOID | F_THRESH | F_OUTF>(
      GG, stream, T0, sT, HH2, Zf, sZ, HH2, nullptr, nullptr, nullptr,
      nullptr, 0, 0, out + o_adjrec, sA, NN, nullptr, 0, 0, T1, sT);

  // (6) xr = bn(relu(z @ Wf1 + bf1))  -> T0 bf16
  launch_gemm<float, bf16_t, HH1, HH2, F_BIAS | F_RELU_FIRST | F_BN | F_OUTB>(
      GG, stream, Zf, sZ, HH2, Wf1t, 0, HH2, bf1, gf, bef,
      nullptr, 0, 0, nullptr, 0, 0, T0, sT, HH1);
  // (7) x_recon = xr @ Wf2 + bf2 -> d_out (f32)
  launch_gemm<bf16_t, bf16_t, FF, HH1, F_BIAS | F_OUTF>(
      GG, stream, T0, sT, HH1, Wf2t, 0, HH1, bf2, nullptr, nullptr,
      nullptr, 0, 0, out + o_xrec, sX, FF);

  // (8) A_hat2 from adj_new (overwrite Abf), then z_tilde = encode(x_recon)
  gcn_norm_k<bf16_t><<<GG, 128, 0, stream>>>(T1, sT, NN, Abf, sA);
  run_encode(out + o_xrec, sX, Abf, T0, T1, T2, B0, B1, out + o_ztilde,
             nullptr, stream);

  // (9) pooling
  maxpool_k<<<GG, 128, 0, stream>>>(Zf, sZ, HH2, nullptr, 0, 0.0f, out + o_zg,
                                    pooled);
  maxpool_k<<<GG, 128, 0, stream>>>(Zf, sZ, HH2, noise, sZ, 0.1f,
                                    pooled + (size_t)GG * HH2, nullptr);
  maxpool_k<<<GG, 128, 0, stream>>>(out + o_ztilde, sZ, HH2, nullptr, 0, 0.0f,
                                    out + o_ztildeg, nullptr);

  // (10) projection head on stacked [512 x 128] pooled matrix (4 row-blocks)
  launch_gemm<float, bf16_t, HH2, HH2, F_BIAS | F_RELU | F_OUTB>(
      4, stream, pooled, (long long)NN * HH2, HH2, Wp1t, 0, HH2, bp1, nullptr,
      nullptr, nullptr, 0, 0, nullptr, 0, 0, T0, (long long)NN * HH2, HH2);
  launch_gemm<bf16_t, bf16_t, HH2, HH2, F_BIAS | F_OUTF>(
      4, stream, T0, (long long)NN * HH2, HH2, Wp2t, 0, HH2, bp2, nullptr,
      nullptr, nullptr, 0, 0, out + o_zgmlp, (long long)NN * HH2, HH2);
}
